// CrossAttentionSALSA3D_65163243815506
// MI455X (gfx1250) — matrix operand order repaired and checked by reading, not compiled, not run on hardware
//
#include <hip/hip_runtime.h>
#include <hip/hip_bf16.h>

// ---------------------------------------------------------------------------
// CrossAttentionSALSA3D for MI455X (gfx1250, wave32, WMMA + TDM)
// B=4, N=1, C=64, IC=16, D=8, H=W=128, k=3 s=2 p=1 -> hp=wp=64, L=4096, E=144
// ---------------------------------------------------------------------------

typedef __attribute__((ext_vector_type(16))) _Float16 v16h;
typedef __attribute__((ext_vector_type(8)))  _Float16 v8h;
typedef __attribute__((ext_vector_type(8)))  float    v8f;
typedef __attribute__((ext_vector_type(4)))  unsigned int u32x4;
typedef __attribute__((ext_vector_type(8)))  int i32x8;
typedef __attribute__((ext_vector_type(4)))  int i32x4;

#define NB    4          // batches
#define CIN   64
#define ICH   16
#define DD    8
#define HH    128
#define WW    128
#define HW    (HH*WW)    // 16384
#define MID   4
#define LTOK  4096       // hp*wp
#define EDIM  144        // IC*3*3
#define EPAD  160        // padded to 5*32 for wmma K
#define KT    32         // keys per flash tile
#define NKT   (LTOK/KT)  // 128 tiles
#define ATT_SCALE 10.0f

// ---------------------------------------------------------------------------
// Kernel 1: 1x1 conv (64->16) on the mid depth slice only.
// which: 0 = g_w(s) -> slot 0 (Q) ; 1 = theta(g) -> slot 2 (V) ;
//        2 = phi(g) -> slot 1 (K)
// ---------------------------------------------------------------------------
__global__ __launch_bounds__(256) void conv_mid_kernel(
    const float* __restrict__ s, const float* __restrict__ g,
    const float* __restrict__ g_w,  const float* __restrict__ g_b,
    const float* __restrict__ th_w, const float* __restrict__ th_b,
    const float* __restrict__ ph_w, const float* __restrict__ ph_b,
    float* __restrict__ convBase)
{
    __shared__ float Wl[ICH * CIN];
    __shared__ float Bl[ICH];
    const int b = blockIdx.z;
    const int which = blockIdx.y;
    const float* w  = (which == 0) ? g_w : (which == 1) ? th_w : ph_w;
    const float* bi = (which == 0) ? g_b : (which == 1) ? th_b : ph_b;
    for (int i = threadIdx.x; i < ICH * CIN; i += 256) Wl[i] = w[i];
    if (threadIdx.x < ICH) Bl[threadIdx.x] = bi[threadIdx.x];
    __syncthreads();

    const int pix = blockIdx.x * 256 + threadIdx.x;
    const float* x = ((which == 0) ? s : g)
                   + ((size_t)b * CIN * DD + MID) * HW + pix;   // + c*DD*HW
    float acc[ICH];
#pragma unroll
    for (int oc = 0; oc < ICH; ++oc) acc[oc] = Bl[oc];
#pragma unroll 4
    for (int c = 0; c < CIN; ++c) {
        float xv = x[(size_t)c * (DD * HW)];
#pragma unroll
        for (int oc = 0; oc < ICH; ++oc)
            acc[oc] = fmaf(Wl[oc * CIN + c], xv, acc[oc]);
    }
    const int slot = (which == 0) ? 0 : (which == 1) ? 2 : 1;
    float* outp = convBase + (size_t)slot * (NB * ICH * HW)
                           + (size_t)b * (ICH * HW);
#pragma unroll
    for (int oc = 0; oc < ICH; ++oc) outp[oc * HW + pix] = acc[oc];
}

// ---------------------------------------------------------------------------
// Kernel 2: extract 3x3/stride2/pad1 patches -> padded f16 (L, 160)
// ---------------------------------------------------------------------------
__global__ __launch_bounds__(256) void pack_kernel(
    const float* __restrict__ convBase, _Float16* __restrict__ qkvBase)
{
    const int b = blockIdx.z, m = blockIdx.y;
    const int idx = blockIdx.x * 256 + threadIdx.x;   // < 4096*160
    const int l = idx / EPAD;
    const int e = idx - l * EPAD;
    float val = 0.0f;
    if (e < EDIM) {
        const int c = e / 9, rem = e % 9, r = rem / 3, cc = rem % 3;
        const int h = 2 * (l >> 6) + r - 1;
        const int w = 2 * (l & 63) + cc - 1;
        if ((unsigned)h < (unsigned)HH && (unsigned)w < (unsigned)WW) {
            const float* cv = convBase + (size_t)m * (NB * ICH * HW)
                                       + (size_t)b * (ICH * HW);
            val = cv[c * HW + h * WW + w];
        }
    }
    qkvBase[(size_t)m * (NB * LTOK * EPAD)
            + ((size_t)b * LTOK + l) * EPAD + e] = (_Float16)val;
}

// ---------------------------------------------------------------------------
// Fragment helpers (per CDNA5 ISA §7.12.2 VGPR layouts, wave32)
// A 16x32 f16: lane = row(lane&15); elems [0..7]=K grp*8.., [8..15]=K 16+grp*8..
// B 32x16 f16: lane = col(lane&15); elems [0..7]=K grp*8.., [8..15]=K 16+grp*8..
// C 16x16 f32: elem r -> (M = r + 8*grp, N = lane&15)
// ---------------------------------------------------------------------------
__device__ inline v16h load_a_frag(const _Float16* p, int ld, int ln, int grp)
{
    const _Float16* r = p + ln * ld;
    v8h lo = *(const v8h*)(r + grp * 8);
    v8h hi = *(const v8h*)(r + 16 + grp * 8);
    v16h a;
#pragma unroll
    for (int i = 0; i < 8; ++i) { a[i] = lo[i]; a[8 + i] = hi[i]; }
    return a;
}

__device__ inline v16h load_b_frag(const _Float16* rowp)
{
    const int g8 = ((threadIdx.x & 31) >> 4) * 8;
    v8h lo = *(const v8h*)(rowp - g8);
    v8h hi = *(const v8h*)(rowp + 16 - g8);
    v16h bfr;
#pragma unroll
    for (int i = 0; i < 8; ++i) { bfr[i] = lo[i]; bfr[8 + i] = hi[i]; }
    return bfr;
}

// ---------------------------------------------------------------------------
// TDM: issue a 2D f16 tile load (KT rows x EPAD elems, row-major -> row-major
// LDS) via the Tensor Data Mover. D# built per CDNA5 ISA §8.3/§8.4.
// ---------------------------------------------------------------------------
__device__ inline void tdm_load_k_tile(const _Float16* gtile, unsigned ldsOff)
{
    unsigned long long ga = (unsigned long long)(const void*)gtile;
    u32x4 g0;
    g0[0] = 1u;                                   // count=1, user mode
    g0[1] = ldsOff;                               // lds_addr (bytes)
    g0[2] = (unsigned)(ga & 0xFFFFFFFFull);       // global_addr[31:0]
    g0[3] = (unsigned)((ga >> 32) & 0x1FFFFFFull) // global_addr[56:32]
          | (2u << 30);                           // type=2 ("image")
    i32x8 g1;
    g1[0] = 0x10000;                 // workgroup_mask=0, data_size=1 (2B)
    g1[1] = (int)(EPAD << 16);       // tensor_dim0[15:0]=160 @bits 63:48
    g1[2] = (int)(LTOK << 16);       // tensor_dim1[15:0]=4096 @bits 95:80
    g1[3] = (int)(EPAD << 16);       // tile_dim0=160 @bits 127:112
    g1[4] = KT;                      // tile_dim1=32, tile_dim2=0
    g1[5] = EPAD;                    // tensor_dim0_stride[31:0]=160
    g1[6] = 0;                       // stride0 hi, dim1_stride lo
    g1[7] = 0;
    i32x4 gz = {0, 0, 0, 0};         // groups 2/3 unused (<=2D tensor)
#if __clang_major__ >= 23
    i32x8 gz8 = {0, 0, 0, 0, 0, 0, 0, 0};
    __builtin_amdgcn_tensor_load_to_lds(g0, g1, gz, gz, gz8, 0);
#else
    __builtin_amdgcn_tensor_load_to_lds(g0, g1, gz, gz, 0);
#endif
}

// ---------------------------------------------------------------------------
// Kernel 3: flash attention. Block = 128 threads = 4 waves; each wave owns 16
// query rows; block covers 64 rows. Loop over 128 key tiles of 32 keys.
// K tile arrives via TDM (tensor_load_to_lds), V is staged transposed by VALU.
// ---------------------------------------------------------------------------
__global__ __launch_bounds__(128) void flash_attn_kernel(
    const _Float16* __restrict__ qh, const _Float16* __restrict__ kh,
    const _Float16* __restrict__ vh, float* __restrict__ zi)
{
    __shared__ __align__(16) _Float16 Ks[KT * EPAD];   // [key][e]   10 KB
    __shared__ __align__(16) _Float16 Vt[EPAD * KT];   // [e][key]   10 KB
    __shared__ __align__(16) _Float16 Ps[4][16 * 32];  // per-wave P  4 KB

    const int b    = blockIdx.y;
    const int tid  = threadIdx.x;
    const int wave = tid >> 5;
    const int lane = tid & 31;
    const int grp  = lane >> 4;
    const int ln   = lane & 15;

    // LDS byte offset of Ks for the TDM descriptor
    const unsigned ksOff =
        (unsigned)(unsigned long long)(__attribute__((address_space(3))) void*)&Ks[0];

    const int qrow0 = blockIdx.x * 64 + wave * 16;
    const _Float16* qbase = qh + ((size_t)b * LTOK + qrow0) * EPAD;

    // Preload the 5 Q A-fragments (16x32 each) for this wave's 16 rows.
    v16h aq[5];
#pragma unroll
    for (int sl = 0; sl < 5; ++sl)
        aq[sl] = load_a_frag(qbase + sl * 32, EPAD, ln, grp);

    float m[8], lsum[8];
    v8f acc[9];
#pragma unroll
    for (int r = 0; r < 8; ++r) { m[r] = -3.0e38f; lsum[r] = 0.0f; }
#pragma unroll
    for (int j = 0; j < 9; ++j) acc[j] = (v8f){};

    const _Float16* kb0 = kh + (size_t)b * LTOK * EPAD;
    const _Float16* vb0 = vh + (size_t)b * LTOK * EPAD;

    for (int kt = 0; kt < NKT; ++kt) {
        const int key0 = kt * KT;
        __syncthreads();                       // prior tile fully consumed

        // --- K tile: async DMA via Tensor Data Mover (wave 0 issues) ---
        if (wave == 0)
            tdm_load_k_tile(kb0 + (size_t)key0 * EPAD, ksOff);

        // --- V tile: stage transposed [e][key] (TDM cannot transpose) ---
        const _Float16* vg = vb0 + (size_t)key0 * EPAD;
#pragma unroll
        for (int it = 0; it < 5; ++it) {
            const int c   = tid + it * 128;    // 640 chunks of 8 halves
            const int row = c / 20;
            const int off = (c % 20) * 8;
            v8h dv = *(const v8h*)(vg + row * EPAD + off);
#pragma unroll
            for (int i = 0; i < 8; ++i) Vt[(off + i) * KT + row] = dv[i];
        }
        if (kt + 1 < NKT) {                    // speculative prefetch -> L2
            __builtin_prefetch(vb0 + (size_t)(key0 + KT) * EPAD + tid * 40, 0, 1);
            __builtin_prefetch(kb0 + (size_t)(key0 + KT) * EPAD + tid * 40, 0, 1);
        }
        if (wave == 0)
            __builtin_amdgcn_s_wait_tensorcnt(0);
        __syncthreads();                       // Ks + Vt visible to all waves

        // ---- S = Q * K^T  (two 16x16 tiles: keys [0..15], [16..31]) ----
        v8f s0 = (v8f){}, s1 = (v8f){};
#pragma unroll
        for (int sl = 0; sl < 5; ++sl) {
            v16h bf0 = load_b_frag(&Ks[ln * EPAD + sl * 32 + grp * 16]);
            v16h bf1 = load_b_frag(&Ks[(16 + ln) * EPAD + sl * 32 + grp * 16]);
            s0 = __builtin_amdgcn_wmma_f32_16x16x32_f16(
                     false, aq[sl], false, bf0, (short)0, s0, false, false);
            s1 = __builtin_amdgcn_wmma_f32_16x16x32_f16(
                     false, aq[sl], false, bf1, (short)0, s1, false, false);
        }

        // ---- online softmax over this key tile ----
        float mx[8], p0[8], p1[8], cf[8], rs[8];
#pragma unroll
        for (int r = 0; r < 8; ++r) {
            float a0 = s0[r] * ATT_SCALE, a1 = s1[r] * ATT_SCALE;
            s0[r] = a0; s1[r] = a1;
            mx[r] = fmaxf(a0, a1);
        }
#pragma unroll
        for (int msk = 1; msk < 16; msk <<= 1)
#pragma unroll
            for (int r = 0; r < 8; ++r)
                mx[r] = fmaxf(mx[r], __shfl_xor(mx[r], msk, 32));
#pragma unroll
        for (int r = 0; r < 8; ++r) {
            float mn = fmaxf(m[r], mx[r]);
            cf[r] = __expf(m[r] - mn);
            m[r]  = mn;
            p0[r] = __expf(s0[r] - mn);
            p1[r] = __expf(s1[r] - mn);
            rs[r] = p0[r] + p1[r];
        }
#pragma unroll
        for (int msk = 1; msk < 16; msk <<= 1)
#pragma unroll
            for (int r = 0; r < 8; ++r)
                rs[r] += __shfl_xor(rs[r], msk, 32);
#pragma unroll
        for (int r = 0; r < 8; ++r) lsum[r] = lsum[r] * cf[r] + rs[r];
#pragma unroll
        for (int j = 0; j < 9; ++j)
#pragma unroll
            for (int r = 0; r < 8; ++r) acc[j][r] *= cf[r];

        // ---- P (16x32 f16) through LDS to get the A layout ----
        _Float16* pa = Ps[wave];
#pragma unroll
        for (int r = 0; r < 8; ++r) {
            pa[(r + 8 * grp) * 32 + ln]      = (_Float16)p0[r];
            pa[(r + 8 * grp) * 32 + 16 + ln] = (_Float16)p1[r];
        }
        v16h aP = load_a_frag(pa, 32, ln, grp);   // same-wave DS ops ordered

        // ---- O += P * V  (9 column tiles = 144 features), loads tripled ----
#pragma unroll
        for (int j = 0; j < 9; j += 3) {
            v16h bv0 = load_b_frag(&Vt[((j + 0) * 16 + ln) * KT + grp * 16]);
            v16h bv1 = load_b_frag(&Vt[((j + 1) * 16 + ln) * KT + grp * 16]);
            v16h bv2 = load_b_frag(&Vt[((j + 2) * 16 + ln) * KT + grp * 16]);
            acc[j + 0] = __builtin_amdgcn_wmma_f32_16x16x32_f16(
                             false, aP, false, bv0, (short)0, acc[j + 0], false, false);
            acc[j + 1] = __builtin_amdgcn_wmma_f32_16x16x32_f16(
                             false, aP, false, bv1, (short)0, acc[j + 1], false, false);
            acc[j + 2] = __builtin_amdgcn_wmma_f32_16x16x32_f16(
                             false, aP, false, bv2, (short)0, acc[j + 2], false, false);
        }
    }

    // ---- epilogue: normalize and write zi (b, row, 144) ----
#pragma unroll
    for (int j = 0; j < 9; ++j) {
#pragma unroll
        for (int r = 0; r < 8; ++r) {
            const int row = qrow0 + r + 8 * grp;
            zi[((size_t)b * LTOK + row) * EDIM + j * 16 + ln] =
                acc[j][r] / lsum[r];
        }
    }
}

// ---------------------------------------------------------------------------
// Kernel 4: fold (overlap-add) + count normalization -> z (b, 16, 128, 128)
// ---------------------------------------------------------------------------
__global__ __launch_bounds__(256) void fold_kernel(
    const float* __restrict__ zi, float* __restrict__ z)
{
    const int b   = blockIdx.y;
    const int idx = blockIdx.x * 256 + threadIdx.x;   // < 16*16384
    const int c   = idx >> 14;
    const int hw  = idx & (HW - 1);
    const int h   = hw >> 7, w = hw & 127;
    const float* zb = zi + (size_t)b * LTOK * EDIM;
    float sum = 0.0f; int cnt = 0;
#pragma unroll
    for (int i = 0; i < 3; ++i) {
        const int hi = h + 1 - i;
        if (hi < 0 || (hi & 1)) continue;
        const int hp = hi >> 1;
        if (hp >= 64) continue;
#pragma unroll
        for (int j = 0; j < 3; ++j) {
            const int wj = w + 1 - j;
            if (wj < 0 || (wj & 1)) continue;
            const int wp = wj >> 1;
            if (wp >= 64) continue;
            sum += zb[(size_t)(hp * 64 + wp) * EDIM + c * 9 + i * 3 + j];
            ++cnt;
        }
    }
    z[(size_t)b * (ICH * HW) + idx] = sum / (float)cnt;
}

// ---------------------------------------------------------------------------
// Kernel 5: final 1x1 conv (16->64) + residual, broadcast over D. Streaming.
// ---------------------------------------------------------------------------
__global__ __launch_bounds__(256) void out_kernel(
    const float* __restrict__ s, const float* __restrict__ z,
    const float* __restrict__ Ww, const float* __restrict__ Wb,
    float* __restrict__ out)
{
    __shared__ float Wl[CIN * ICH];
    __shared__ float Bl[CIN];
    for (int i = threadIdx.x; i < CIN * ICH; i += 256) Wl[i] = Ww[i];
    if (threadIdx.x < CIN) Bl[threadIdx.x] = Wb[threadIdx.x];
    __syncthreads();

    const int b   = blockIdx.y;
    const int pix = blockIdx.x * 256 + threadIdx.x;
    float zv[ICH];
#pragma unroll
    for (int c = 0; c < ICH; ++c)
        zv[c] = z[(size_t)b * (ICH * HW) + c * HW + pix];

#pragma unroll 2
    for (int co = 0; co < CIN; ++co) {
        float y = Bl[co];
#pragma unroll
        for (int c = 0; c < ICH; ++c) y = fmaf(Wl[co * ICH + c], zv[c], y);
        const size_t base = ((size_t)b * CIN + co) * (DD * HW) + pix;
#pragma unroll
        for (int d = 0; d < DD; ++d)
            out[base + (size_t)d * HW] = s[base + (size_t)d * HW] + y;
    }
}

// ---------------------------------------------------------------------------
// Host launcher
// ---------------------------------------------------------------------------
extern "C" void kernel_launch(void* const* d_in, const int* in_sizes, int n_in,
                              void* d_out, int out_size, void* d_ws, size_t ws_size,
                              hipStream_t stream) {
    const float* s       = (const float*)d_in[0];
    const float* g       = (const float*)d_in[1];
    const float* g_w     = (const float*)d_in[2];
    const float* g_bias  = (const float*)d_in[3];
    const float* theta_w = (const float*)d_in[4];
    const float* theta_b = (const float*)d_in[5];
    const float* phi_w   = (const float*)d_in[6];
    const float* phi_b   = (const float*)d_in[7];
    const float* W_w     = (const float*)d_in[8];
    const float* W_bias  = (const float*)d_in[9];
    float* out = (float*)d_out;

    // Workspace layout (bytes): conv[3] | zi | z | qkv f16 [3]   (~40 MiB)
    const size_t CONV_BYTES = (size_t)3 * NB * ICH * HW * sizeof(float);
    const size_t ZI_BYTES   = (size_t)NB * LTOK * EDIM * sizeof(float);
    const size_t Z_BYTES    = (size_t)NB * ICH * HW * sizeof(float);
    char* wsb = (char*)d_ws;
    float*     convBase = (float*)wsb;
    float*     zi       = (float*)(wsb + CONV_BYTES);
    float*     z        = (float*)(wsb + CONV_BYTES + ZI_BYTES);
    _Float16*  qkvBase  = (_Float16*)(wsb + CONV_BYTES + ZI_BYTES + Z_BYTES);
    const size_t QKV_ELEMS = (size_t)NB * LTOK * EPAD;
    _Float16* qh = qkvBase;
    _Float16* kh = qkvBase + QKV_ELEMS;
    _Float16* vh = qkvBase + 2 * QKV_ELEMS;

    conv_mid_kernel<<<dim3(HW / 256, 3, NB), 256, 0, stream>>>(
        s, g, g_w, g_bias, theta_w, theta_b, phi_w, phi_b, convBase);

    pack_kernel<<<dim3((LTOK * EPAD) / 256, 3, NB), 256, 0, stream>>>(
        convBase, qkvBase);

    flash_attn_kernel<<<dim3(LTOK / 64, NB), 128, 0, stream>>>(qh, kh, vh, zi);

    fold_kernel<<<dim3((ICH * HW) / 256, NB), 256, 0, stream>>>(zi, z);

    out_kernel<<<dim3(HW / 256, NB), 256, 0, stream>>>(s, z, W_w, W_bias, out);
}
